// MambaBlock_22763326669194
// MI455X (gfx1250) — matrix operand order repaired and checked by reading, not compiled, not run on hardware
//
#include <hip/hip_runtime.h>
#include <hip/hip_bf16.h>

typedef __attribute__((ext_vector_type(16))) __bf16 v16bf;
typedef __attribute__((ext_vector_type(8)))  __bf16 v8bf;
typedef __attribute__((ext_vector_type(8)))  float  v8f;

// ---- problem constants ----
#define BB       2
#define LL       4096
#define DMODEL   1024
#define DINNER   2048
#define DSTATE   16
#define DCONV    4
#define DTRANK   64
#define MROWS    (BB * LL)          // 8192
#define XZ_LD    (2 * DINNER)       // 4096

// ---- GEMM tiling ----
#define BLK_M 128
#define BLK_N 128
#define BLK_K 32
#define TLD   40                    // LDS row pitch: 40 bf16 = 80B (16B aligned, conflict-free)

// async global->LDS 16B copy (CDNA5, ASYNCcnt-tracked)
#define ASYNC_B128(ldsaddr, gaddr)                                             \
    asm volatile("global_load_async_to_lds_b128 %0, %1, off"                   \
                 :: "v"(ldsaddr), "v"(gaddr) : "memory")
#define WAIT_ASYNC0() asm volatile("s_wait_asynccnt 0x0" ::: "memory")

__device__ __forceinline__ unsigned lds_off(const void* p) {
    return (unsigned)(uintptr_t)p;   // LDS aperture: addr[31:0] == wave-relative LDS offset
}

// split one fp32 matrix (possibly strided rows) into compact bf16 hi/lo planes
__global__ __launch_bounds__(256) void cvt_split_kernel(
    const float* __restrict__ src, int ld, int cols, int total,
    __bf16* __restrict__ hi, __bf16* __restrict__ lo)
{
    const int i = blockIdx.x * 256 + threadIdx.x;
    if (i >= total) return;
    const int row = i / cols;
    const int col = i - row * cols;
    const float v = src[(size_t)row * ld + col];
    const __bf16 h = (__bf16)v;
    hi[i] = h;
    lo[i] = (__bf16)(v - (float)h);
}

// A frag (16-bit A 16x32, §7.12.2): lane half kh -> elements 0..7 = K kh*8+0..7,
// elements 8..15 = K 16+kh*8+0..7.  p points at row base + kh*8.
__device__ __forceinline__ v16bf frag_a(const __bf16* p) {
    v8bf x0 = *(const v8bf*)(p);
    v8bf x1 = *(const v8bf*)(p + 16);
    return __builtin_shufflevector(x0, x1, 0,1,2,3,4,5,6,7,8,9,10,11,12,13,14,15);
}
// B frag (32x16): elements 0..7 = K kh*8+0..7, elements 8..15 = K 16+kh*8+0..7, N = lane&15.
__device__ __forceinline__ v16bf frag_b(const __bf16* p) {
    v8bf x0 = *(const v8bf*)(p);
    v8bf x1 = *(const v8bf*)(p + 16);
    return __builtin_shufflevector(x0, x1, 0,1,2,3,4,5,6,7,8,9,10,11,12,13,14,15);
}

// C[M,N] = A[M,K] @ Bw[N,K]^T via bf16 hi/lo 3-term split, fp32 accumulate.
// Block tile 128x128, 8 waves, wave tile 32x64 (24 wmma/K-step).
// Double-buffered LDS fed by global_load_async_to_lds_b128 (one barrier/K-step).
// mode 0: plain store; mode 1: softplus(C + bias[n]) (dt_proj epilogue)
__global__ __launch_bounds__(256) void gemm_bf16x3(
    const __bf16* __restrict__ Ahi, const __bf16* __restrict__ Alo,
    const __bf16* __restrict__ Bhi, const __bf16* __restrict__ Blo,
    float* __restrict__ C, int ldc, int N, int K, int mode,
    const float* __restrict__ bias)
{
    __shared__ __attribute__((aligned(16))) __bf16 As[2][2][BLK_M][TLD]; // [buf][plane][m][k]
    __shared__ __attribute__((aligned(16))) __bf16 Bs[2][2][BLK_N][TLD]; // [buf][plane][n][k]

    const int tid    = threadIdx.x;
    const int lane   = tid & 31;
    const int wid    = tid >> 5;
    const int wave_m = wid & 3;     // 4 waves x 32 rows
    const int wave_n = wid >> 2;    // 2 waves x 64 cols
    const int m0 = blockIdx.y * BLK_M;
    const int n0 = blockIdx.x * BLK_N;
    const int r  = lane & 15;
    const int kh = lane >> 4;

    // staging map: e = tid + j*256, row = e>>2 (0..127), c8 = e&3 (8 bf16 each)
    const int srow = tid >> 2;
    const int sc8  = tid & 3;

    v8f acc[2][4] = {};
    const v8bf vz = {};

    auto stage = [&](int buf, int k0) {
        #pragma unroll
        for (int j = 0; j < 2; ++j) {
            const int row = srow + j * 64;
            const size_t offA = (size_t)(m0 + row) * K + k0 + sc8 * 8;
            ASYNC_B128(lds_off(&As[buf][0][row][sc8 * 8]), (unsigned long long)(uintptr_t)(Ahi + offA));
            ASYNC_B128(lds_off(&As[buf][1][row][sc8 * 8]), (unsigned long long)(uintptr_t)(Alo + offA));
            if (n0 + row < N) {
                const size_t offB = (size_t)(n0 + row) * K + k0 + sc8 * 8;
                ASYNC_B128(lds_off(&Bs[buf][0][row][sc8 * 8]), (unsigned long long)(uintptr_t)(Bhi + offB));
                ASYNC_B128(lds_off(&Bs[buf][1][row][sc8 * 8]), (unsigned long long)(uintptr_t)(Blo + offB));
            } else {
                *(v8bf*)&Bs[buf][0][row][sc8 * 8] = vz;
                *(v8bf*)&Bs[buf][1][row][sc8 * 8] = vz;
            }
        }
    };

    stage(0, 0);                     // prologue: DMA tile 0 into buffer 0
    int buf = 0;

    for (int k0 = 0; k0 < K; k0 += BLK_K) {
        WAIT_ASYNC0();               // own wave's DMAs done...
        __syncthreads();             // ...all waves' slices resident (also waits DScnt)

        if (k0 + BLK_K < K) stage(buf ^ 1, k0 + BLK_K);   // overlap next DMA with compute

        v16bf ah[2], al[2], bh[4], bl[4];
        #pragma unroll
        for (int mt = 0; mt < 2; ++mt) {
            ah[mt] = frag_a(&As[buf][0][wave_m * 32 + mt * 16 + r][kh * 8]);
            al[mt] = frag_a(&As[buf][1][wave_m * 32 + mt * 16 + r][kh * 8]);
        }
        #pragma unroll
        for (int nt = 0; nt < 4; ++nt) {
            bh[nt] = frag_b(&Bs[buf][0][wave_n * 64 + nt * 16 + r][kh * 8]);
            bl[nt] = frag_b(&Bs[buf][1][wave_n * 64 + nt * 16 + r][kh * 8]);
        }
        #pragma unroll
        for (int mt = 0; mt < 2; ++mt)
            #pragma unroll
            for (int nt = 0; nt < 4; ++nt) {
                acc[mt][nt] = __builtin_amdgcn_wmma_f32_16x16x32_bf16(false, ah[mt], false, bh[nt], (short)0, acc[mt][nt], false, false);
                acc[mt][nt] = __builtin_amdgcn_wmma_f32_16x16x32_bf16(false, ah[mt], false, bl[nt], (short)0, acc[mt][nt], false, false);
                acc[mt][nt] = __builtin_amdgcn_wmma_f32_16x16x32_bf16(false, al[mt], false, bh[nt], (short)0, acc[mt][nt], false, false);
            }
        buf ^= 1;
    }

    // epilogue: C layout §7.12.2 — VGPR v = row v + 8*kh, col = lane&15
    #pragma unroll
    for (int mt = 0; mt < 2; ++mt) {
        #pragma unroll
        for (int nt = 0; nt < 4; ++nt) {
            const int colg = n0 + wave_n * 64 + nt * 16 + r;
            if (colg >= N) continue;
            const int rowb = m0 + wave_m * 32 + mt * 16 + kh * 8;
            #pragma unroll
            for (int v = 0; v < 8; ++v) {
                float cval = acc[mt][nt][v];
                if (mode == 1) {
                    const float xv = cval + bias[colg];
                    cval = (xv > 20.0f) ? xv : log1pf(__expf(xv));
                }
                C[(size_t)(rowb + v) * ldc + colg] = cval;
            }
        }
    }
}

// causal depthwise conv1d (pad left 3) + bias + SiLU, fused with bf16 hi/lo split.
// x lives in cols [0,DINNER) of xz; output planes are GEMM3's A and the scan's u.
__global__ __launch_bounds__(256) void conv_silu_kernel(
    const float* __restrict__ xz, const float* __restrict__ conv_w,
    const float* __restrict__ conv_b, __bf16* __restrict__ xhi,
    __bf16* __restrict__ xlo)
{
    const int idx = blockIdx.x * 256 + threadIdx.x;        // over B*L*DINNER
    const int d = idx & (DINNER - 1);
    const int l = (idx >> 11) & (LL - 1);
    const int b = idx >> 23;

    float acc = conv_b[d];
    #pragma unroll
    for (int j = 0; j < DCONV; ++j) {
        const int lt = l - (DCONV - 1) + j;
        if (lt >= 0)
            acc += conv_w[d * DCONV + j] * xz[(size_t)(b * LL + lt) * XZ_LD + d];
    }
    const float s = acc / (1.0f + __expf(-acc));           // SiLU
    const size_t i = (size_t)(b * LL + l) * DINNER + d;
    const __bf16 h = (__bf16)s;
    xhi[i] = h;
    xlo[i] = (__bf16)(s - (float)h);
}

// selective scan: 16 lanes per channel (wave32 -> 2 channels/wave).
// delta sits in cols [0,DINNER) of xz, z in cols [DINNER,2*DINNER).
// u is reconstructed from the x hi/lo planes; gated y is written back (bf16
// hi/lo split) IN PLACE over the x planes -> directly GEMM6's A operand.
// (in-place safe: the stored y depends on the loaded u, so the load completes
//  before the store issues; no other wave touches this (b,d) column.)
__global__ __launch_bounds__(256) void scan_kernel(
    const float* __restrict__ xz, __bf16* __restrict__ yhi,
    __bf16* __restrict__ ylo, const float* __restrict__ xp,
    const float* __restrict__ A_log, const float* __restrict__ Dvec)
{
    const int gid  = blockIdx.x * 256 + threadIdx.x;
    const int lane = threadIdx.x & 31;
    const int wave = gid >> 5;
    const int half = lane >> 4;          // channel within wave
    const int n    = lane & 15;          // state index

    const int c = wave * 2 + half;       // 0 .. B*DINNER-1
    const int d = c & (DINNER - 1);
    const int b = c >> 11;

    const float a  = -__expf(A_log[d * DSTATE + n]);
    const float Dv = Dvec[d];

    float h = 0.0f;
    for (int t = 0; t < LL; ++t) {
        const size_t row = (size_t)(b * LL + t);
        const size_t ui  = row * DINNER + d;
        const float dlt = xz[row * XZ_LD + d];
        const float u   = (float)yhi[ui] + (float)ylo[ui];
        const float Bv  = xp[row * (DTRANK + 2 * DSTATE) + DTRANK + n];
        const float Cv  = xp[row * (DTRANK + 2 * DSTATE) + DTRANK + DSTATE + n];

        h = h * __expf(dlt * a) + (dlt * u) * Bv;

        float p = h * Cv;                               // reduce 16 states
        p += __shfl_xor(p, 1, 32);
        p += __shfl_xor(p, 2, 32);
        p += __shfl_xor(p, 4, 32);
        p += __shfl_xor(p, 8, 32);

        if (n == 0) {
            const float z = xz[row * XZ_LD + DINNER + d];
            const float y = (p + Dv * u) * (z / (1.0f + __expf(-z)));
            const __bf16 hy = (__bf16)y;
            yhi[ui] = hy;                               // overwrite x planes
            ylo[ui] = (__bf16)(y - (float)hy);
        }
    }
}

static inline void cvt_split(const float* src, int ld, int rows, int cols,
                             __bf16* hi, __bf16* lo, hipStream_t s) {
    const int total = rows * cols;
    cvt_split_kernel<<<(total + 255) / 256, 256, 0, s>>>(src, ld, cols, total, hi, lo);
}

extern "C" void kernel_launch(void* const* d_in, const int* in_sizes, int n_in,
                              void* d_out, int out_size, void* d_ws, size_t ws_size,
                              hipStream_t stream) {
    const float* hidden     = (const float*)d_in[0];
    const float* in_proj_w  = (const float*)d_in[1];
    const float* conv_w     = (const float*)d_in[2];
    const float* conv_b     = (const float*)d_in[3];
    const float* x_proj_w   = (const float*)d_in[4];
    const float* dt_w       = (const float*)d_in[5];
    const float* dt_b       = (const float*)d_in[6];
    const float* A_log      = (const float*)d_in[7];
    const float* Dvec       = (const float*)d_in[8];
    const float* out_proj_w = (const float*)d_in[9];
    float* out = (float*)d_out;

    // ---- workspace carve-up ----
    float*  xz    = (float*)d_ws;                           // 8192 x 4096 f32
    float*  xp    = xz + (size_t)MROWS * XZ_LD;             // 8192 x 96 f32
    __bf16* Ahi   = (__bf16*)(xp + (size_t)MROWS * 96);     // 8192 x 2048 bf16 (hidden -> x -> y planes)
    __bf16* Alo   = Ahi + (size_t)MROWS * DINNER;
    __bf16* Bhi   = Alo + (size_t)MROWS * DINNER;           // <= 4096 x 1024 bf16 (weight planes)
    __bf16* Blo   = Bhi + (size_t)(2 * DINNER) * DMODEL;
    __bf16* DThi  = Blo + (size_t)(2 * DINNER) * DMODEL;    // 8192 x 64 bf16 (dt planes)
    __bf16* DTlo  = DThi + (size_t)MROWS * DTRANK;

    const dim3 blk(256);
    const int mTiles = MROWS / BLK_M;                       // 64

    // 1) xz = H @ in_proj_w^T        (N=4096, K=1024)
    cvt_split(hidden, DMODEL, MROWS, DMODEL, Ahi, Alo, stream);
    cvt_split(in_proj_w, DMODEL, 2 * DINNER, DMODEL, Bhi, Blo, stream);
    gemm_bf16x3<<<dim3(2 * DINNER / BLK_N, mTiles), blk, 0, stream>>>(
        Ahi, Alo, Bhi, Blo, xz, XZ_LD, 2 * DINNER, DMODEL, 0, nullptr);

    // 2) depthwise causal conv + SiLU -> x hi/lo planes (overwrites hidden planes)
    conv_silu_kernel<<<(MROWS * DINNER) / 256, blk, 0, stream>>>(xz, conv_w, conv_b, Ahi, Alo);

    // 3) xp = silu(conv(x)) @ x_proj_w^T   (N=96, K=2048), A = fused x planes
    cvt_split(x_proj_w, DINNER, DTRANK + 2 * DSTATE, DINNER, Bhi, Blo, stream);
    gemm_bf16x3<<<dim3((DTRANK + 2 * DSTATE + BLK_N - 1) / BLK_N, mTiles), blk, 0, stream>>>(
        Ahi, Alo, Bhi, Blo, xp, DTRANK + 2 * DSTATE, DTRANK + 2 * DSTATE, DINNER, 0, nullptr);

    // 4) delta = softplus(dt @ dt_w^T + dt_b) -> overwrite x half of xz (ldc=4096)
    cvt_split(xp, DTRANK + 2 * DSTATE, MROWS, DTRANK, DThi, DTlo, stream);
    cvt_split(dt_w, DTRANK, DINNER, DTRANK, Bhi, Blo, stream);
    gemm_bf16x3<<<dim3(DINNER / BLK_N, mTiles), blk, 0, stream>>>(
        DThi, DTlo, Bhi, Blo, xz, XZ_LD, DINNER, DTRANK, 1, dt_b);

    // 5) selective scan (+ D skip + SiLU(z) gating); y planes replace x planes
    scan_kernel<<<(BB * DINNER / 2) * 32 / 256, blk, 0, stream>>>(xz, Ahi, Alo, xp, A_log, Dvec);

    // 6) out = y @ out_proj_w^T      (N=1024, K=2048), A = fused y planes
    cvt_split(out_proj_w, DINNER, DMODEL, DINNER, Bhi, Blo, stream);
    gemm_bf16x3<<<dim3(DMODEL / BLK_N, mTiles), blk, 0, stream>>>(
        Ahi, Alo, Bhi, Blo, out, DMODEL, DMODEL, DINNER, 0, nullptr);
}
